// Starformer_11854109737562
// MI455X (gfx1250) — matrix operand order repaired and checked by reading, not compiled, not run on hardware
//
#include <hip/hip_runtime.h>
#include <hip/hip_bf16.h>
#include <math.h>

// ---------------------------------------------------------------------------
// Starformer forward for MI455X (gfx1250, wave32, WMMA).
// Big contractions: v_wmma_f32_16x16x32_f16, async-LDS B staging, double buffer.
// ---------------------------------------------------------------------------

typedef __attribute__((ext_vector_type(16))) _Float16 v16h;
typedef __attribute__((ext_vector_type(8)))  float    v8f;
typedef __attribute__((ext_vector_type(4)))  unsigned int u32x4;

union Frag { u32x4 u[2]; v16h h; };

#define BB   64
#define TT   64
#define CC   64
#define VV   18
#define ID   128     // local width
#define LHH  8       // local heads (dh=16)
#define DD   512     // global width
#define GHH  8       // global heads (dh=64)
#define NLL  6
#define PP   65      // C+1 tokens per local sequence
#define BT   (BB*TT)        // 4096
#define R1   (BT*PP)        // 266240 local rows
#define R2   (BB*2*TT)      // 8192 global rows (seq len 128 per batch)

// GEMM tile geometry (shared between kernel and host launch)
#define G_BM 128
#define G_BN 128
#define G_BK 32
#define G_LDT 40                              // padded halves per tile row
#define G_TILE_BYTES (G_BM * G_LDT * 2)       // 10240 B per buffer
#define G_SA_BYTES   (2 * G_TILE_BYTES)       // double-buffered A
#define G_SB_OFF     G_SA_BYTES
#define G_SMEM_BYTES (4 * G_TILE_BYTES)       // 40960 B total dynamic LDS

// ---------------------------------------------------------------------------
// Generic GEMM: C[M,N] = act( A[M,K](f32) @ Wt[N,K](f16, pre-transposed)
//                             + bias[N] (+ Res) )
// REQUIRES K % 32 == 0 (true for all call sites: 128/512/2048/8320).
// 256 threads = 8 waves; block tile 128x128, K-step 32; wave tile 32x64.
// B tiles staged with global_load_async_to_lds_b128 (ASYNCcnt), A tiles with
// batched global_load_b128 + f32->f16 convert + ds_store; double-buffered so
// next-tile staging overlaps WMMA on the current tile.
// ACT: 0 = none, 1 = exact GELU, 2 = tanh
// ---------------------------------------------------------------------------
template<int ACT>
__global__ __launch_bounds__(256)
void gemm_f16w_kernel(const float* __restrict__ A, const _Float16* __restrict__ Wt,
                      const float* __restrict__ bias, const float* __restrict__ Res,
                      float* __restrict__ Cout, int M, int N, int K)
{
  extern __shared__ char smem[];                       // dynamic LDS, offset 0
  _Float16* sA = (_Float16*)smem;                      // [2][128][G_LDT]
  _Float16* sB = (_Float16*)(smem + G_SB_OFF);         // [2][128][G_LDT] (N-major)

  const int tid  = threadIdx.x;
  const int lane = tid & 31;
  const int w    = tid >> 5;
  const int wm   = w >> 1;        // 0..3
  const int wn   = w & 1;         // 0..1
  const int m0   = blockIdx.y * G_BM;
  const int n0   = blockIdx.x * G_BN;
  const int KT   = K / G_BK;

  // ---- branch-free staging helpers (clamped indices; OOB masked at store) --
  // A: issue all 4 global loads first (one loadcnt wait), then convert+store.
  auto stageA = [&](int kt, int buf) {
    const int k0 = kt * G_BK;
    const int base = buf * (G_BM * G_LDT);
    float4 vals[4];
#pragma unroll
    for (int it = 0; it < 4; ++it) {
      int id = tid + it * 256;            // 0..1023 float4 slots
      int r  = id >> 3;                   // tile row
      int c4 = (id & 7) << 2;             // col (multiple of 4)
      int gm = m0 + r; gm = gm < M ? gm : M - 1;
      vals[it] = *(const float4*)(A + (size_t)gm * K + k0 + c4);
    }
#pragma unroll
    for (int it = 0; it < 4; ++it) {
      int id = tid + it * 256;
      int r  = id >> 3;
      int c4 = (id & 7) << 2;
      _Float16* dst = sA + base + r * G_LDT + c4;
      dst[0] = (_Float16)vals[it].x;
      dst[1] = (_Float16)vals[it].y;
      dst[2] = (_Float16)vals[it].z;
      dst[3] = (_Float16)vals[it].w;
    }
  };
  auto stageB = [&](int kt, int buf) {
    const int k0 = kt * G_BK;
#pragma unroll
    for (int it = 0; it < 2; ++it) {
      int id = tid + it * 256;            // 0..511 16-byte chunks
      int nn = id >> 2;                   // tile col (output n), 0..127
      int ck = id & 3;                    // chunk within 32-half row
      int gn = n0 + nn; gn = gn < N ? gn : N - 1;
      const _Float16* src = Wt + (size_t)gn * K + k0 + ck * 8;
      unsigned dstoff = (unsigned)(G_SB_OFF +
                        (buf * (G_BM * G_LDT) + nn * G_LDT + ck * 8) * 2);
      asm volatile("global_load_async_to_lds_b128 %0, %1, off"
                   :: "v"(dstoff), "v"(src) : "memory");
    }
  };

  const v8f vzero = {0.f,0.f,0.f,0.f,0.f,0.f,0.f,0.f};
  v8f acc[2][4];
#pragma unroll
  for (int i = 0; i < 2; ++i)
#pragma unroll
    for (int j = 0; j < 4; ++j) acc[i][j] = vzero;

  // prologue: stage tile 0 into buffer 0
  stageB(0, 0);
  stageA(0, 0);
  asm volatile("s_wait_asynccnt 0x0" ::: "memory");
  __syncthreads();

  for (int kt = 0; kt < KT; ++kt) {
    const int cur = kt & 1;
    if (kt + 1 < KT) {                    // overlap next-tile staging with WMMA
      stageB(kt + 1, cur ^ 1);
      stageA(kt + 1, cur ^ 1);
    }

    // ---- fragments per documented CDNA5 16-bit WMMA striping ---------------
    // A 16x32: lane<16 -> K 0..7 & 16..23 ; lane>=16 -> K 8..15 & 24..31
    // B 32x16: lane<16 -> K 0..7 & 16..23 of col ; lane>=16 -> K 8..15 & 24..31 of col
    Frag af[2], bf[4];
    {
      const _Float16* a = sA + cur * (G_BM * G_LDT);
      const _Float16* b = sB + cur * (G_BM * G_LDT);
      const int kb = (lane >> 4) * 8;
#pragma unroll
      for (int mi = 0; mi < 2; ++mi) {
        int row = wm * 32 + mi * 16 + (lane & 15);
        af[mi].u[0] = *(const u32x4*)(a + row * G_LDT + kb);
        af[mi].u[1] = *(const u32x4*)(a + row * G_LDT + kb + 16);
      }
      const int kb2 = (lane >> 4) * 8;
#pragma unroll
      for (int ni = 0; ni < 4; ++ni) {
        int nr = wn * 64 + ni * 16 + (lane & 15);
        bf[ni].u[0] = *(const u32x4*)(b + nr * G_LDT + kb2);
        bf[ni].u[1] = *(const u32x4*)(b + nr * G_LDT + kb2 + 16);
      }
    }

#pragma unroll
    for (int mi = 0; mi < 2; ++mi)
#pragma unroll
      for (int ni = 0; ni < 4; ++ni)
        acc[mi][ni] = __builtin_amdgcn_wmma_f32_16x16x32_f16(
            false, af[mi].h, false, bf[ni].h, (short)0, acc[mi][ni], false, false);

    asm volatile("s_wait_asynccnt 0x0" ::: "memory");
    __syncthreads();
  }

  // ---- epilogue: C/D layout => lanes 0-15: M=r, N=lane; 16-31: M=r+8 -------
#pragma unroll
  for (int mi = 0; mi < 2; ++mi)
#pragma unroll
    for (int ni = 0; ni < 4; ++ni)
#pragma unroll
      for (int r = 0; r < 8; ++r) {
        int m = m0 + wm * 32 + mi * 16 + r + ((lane >= 16) ? 8 : 0);
        int n = n0 + wn * 64 + ni * 16 + (lane & 15);
        if (m < M && n < N) {
          float vx = acc[mi][ni][r] + bias[n];
          if (Res) vx += Res[(size_t)m * N + n];
          if (ACT == 1) vx = 0.5f * vx * (1.0f + erff(vx * 0.70710678118f));
          else if (ACT == 2) vx = tanhf(vx);
          Cout[(size_t)m * N + n] = vx;
        }
      }
}

// ---------------------------------------------------------------------------
// Weight prep: Wt[n][k] = (f16) W[k][n]   (LDS-tiled transpose + convert)
// ---------------------------------------------------------------------------
__global__ __launch_bounds__(256)
void transpose_cvt_kernel(const float* __restrict__ W, _Float16* __restrict__ Wt,
                          int K, int N)
{
  __shared__ float tile[32][33];
  const int kb = blockIdx.y * 32, nb = blockIdx.x * 32;
  const int tx = threadIdx.x & 31, ty = threadIdx.x >> 5;   // 32x8
#pragma unroll
  for (int i = 0; i < 32; i += 8) {
    int k = kb + ty + i, n = nb + tx;
    tile[ty + i][tx] = (k < K && n < N) ? W[(size_t)k * N + n] : 0.f;
  }
  __syncthreads();
#pragma unroll
  for (int i = 0; i < 32; i += 8) {
    int n = nb + ty + i, k = kb + tx;
    if (n < N && k < K) Wt[(size_t)n * K + k] = (_Float16)tile[tx][ty + i];
  }
}

// ---------------------------------------------------------------------------
// LayerNorm: one block per row.
// ---------------------------------------------------------------------------
__global__ __launch_bounds__(256)
void layernorm_kernel(const float* __restrict__ X, const float* __restrict__ g,
                      const float* __restrict__ b, float* __restrict__ Y, int dim)
{
  __shared__ float red[256];
  const int row = blockIdx.x;
  const float* x = X + (size_t)row * dim;
  float s = 0.f;
  for (int i = threadIdx.x; i < dim; i += 256) s += x[i];
  red[threadIdx.x] = s; __syncthreads();
  for (int st = 128; st > 0; st >>= 1) {
    if (threadIdx.x < st) red[threadIdx.x] += red[threadIdx.x + st];
    __syncthreads();
  }
  const float mean = red[0] / (float)dim;
  __syncthreads();
  float vv = 0.f;
  for (int i = threadIdx.x; i < dim; i += 256) { float d = x[i] - mean; vv += d * d; }
  red[threadIdx.x] = vv; __syncthreads();
  for (int st = 128; st > 0; st >>= 1) {
    if (threadIdx.x < st) red[threadIdx.x] += red[threadIdx.x + st];
    __syncthreads();
  }
  const float rstd = rsqrtf(red[0] / (float)dim + 1e-5f);
  float* y = Y + (size_t)row * dim;
  for (int i = threadIdx.x; i < dim; i += 256)
    y[i] = (x[i] - mean) * rstd * g[i] + b[i];
}

// ---------------------------------------------------------------------------
// Embeddings
// ---------------------------------------------------------------------------
__global__ void embed_local_kernel(const float* __restrict__ states, const float* __restrict__ actions,
                                   const float* __restrict__ actW, const float* __restrict__ actB,
                                   const float* __restrict__ patchW, const float* __restrict__ patchB,
                                   const float* __restrict__ spatial, float* __restrict__ out)
{
  size_t idx = (size_t)blockIdx.x * blockDim.x + threadIdx.x;
  const size_t total = (size_t)BT * PP * ID;
  if (idx >= total) return;
  int ch = (int)(idx & (ID - 1));
  size_t rp = idx >> 7;
  int p  = (int)(rp % PP);
  size_t bt = rp / PP;
  float val;
  if (p == 0) {
    float acc = actB[ch];
    for (int a = 0; a < VV; ++a) acc += actions[bt * VV + a] * actW[a * ID + ch];
    val = acc;
  } else {
    int c = p - 1;
    val = states[bt * CC + c] * patchW[ch] + patchB[ch] + spatial[c * ID + ch];
  }
  out[idx] = val;
}

__global__ void embed_glob_kernel(const float* __restrict__ states, const float* __restrict__ vecW,
                                  const float* __restrict__ vecB, const float* __restrict__ temporal,
                                  float* __restrict__ out)
{
  size_t idx = (size_t)blockIdx.x * blockDim.x + threadIdx.x;
  const size_t total = (size_t)BT * DD;
  if (idx >= total) return;
  int n = (int)(idx & (DD - 1));
  size_t bt = idx >> 9;
  int t = (int)(bt & (TT - 1));
  float acc = vecB[n] + temporal[t * DD + n];
  for (int c = 0; c < CC; ++c) acc += states[bt * CC + c] * vecW[c * DD + n];
  out[idx] = acc;
}

// ---------------------------------------------------------------------------
// Local attention: one block per (sequence, head). N=65, dh=16, no mask.
// ---------------------------------------------------------------------------
__global__ __launch_bounds__(128)
void attn_local_kernel(const float* __restrict__ Q, const float* __restrict__ K,
                       const float* __restrict__ V, float* __restrict__ Y)
{
  __shared__ float sq[PP][16], sk[PP][16], sv[PP][16];
  __shared__ float ss[PP][PP + 1];
  const int sh  = blockIdx.x;
  const int seq = sh / LHH;
  const int h   = sh % LHH;
  const int tid = threadIdx.x;
  const size_t rowbase = (size_t)seq * PP;

  for (int i = tid; i < PP * 16; i += 128) {
    int r = i >> 4, c = i & 15;
    size_t g = (rowbase + r) * ID + h * 16 + c;
    sq[r][c] = Q[g]; sk[r][c] = K[g]; sv[r][c] = V[g];
  }
  __syncthreads();

  for (int i = tid; i < PP * PP; i += 128) {
    int n = i / PP, m = i % PP;
    float acc = 0.f;
#pragma unroll
    for (int d = 0; d < 16; ++d) acc += sq[n][d] * sk[m][d];
    ss[n][m] = acc * 0.25f;                  // 1/sqrt(16)
  }
  __syncthreads();

  if (tid < PP) {
    float mx = -3.0e38f;
    for (int m = 0; m < PP; ++m) mx = fmaxf(mx, ss[tid][m]);
    float sum = 0.f;
    for (int m = 0; m < PP; ++m) { float e = expf(ss[tid][m] - mx); ss[tid][m] = e; sum += e; }
    float inv = 1.0f / sum;
    for (int m = 0; m < PP; ++m) ss[tid][m] *= inv;
  }
  __syncthreads();

  for (int i = tid; i < PP * 16; i += 128) {
    int n = i >> 4, c = i & 15;
    float acc = 0.f;
    for (int m = 0; m < PP; ++m) acc += ss[n][m] * sv[m][c];
    Y[(rowbase + n) * ID + h * 16 + c] = acc;
  }
}

// ---------------------------------------------------------------------------
// Global attention: one block per (batch, head, 16-row query chunk).
// N=128, dh=64, mask = tril + (even row n attends to (n-1) mod 128).
// ---------------------------------------------------------------------------
__global__ __launch_bounds__(256)
void attn_glob_kernel(const float* __restrict__ Q, const float* __restrict__ K,
                      const float* __restrict__ V, float* __restrict__ Y)
{
  __shared__ _Float16 sk[128][64];
  __shared__ _Float16 sv[128][64];
  __shared__ float ss[16][132];
  const int id    = blockIdx.x;
  const int chunk = id & 7;
  const int bh    = id >> 3;
  const int b     = bh / GHH;
  const int h     = bh % GHH;
  const int tid   = threadIdx.x;
  const size_t rowbase = (size_t)b * 128;

  for (int i = tid; i < 128 * 64; i += 256) {
    int r = i >> 6, c = i & 63;
    size_t g = (rowbase + r) * DD + h * 64 + c;
    sk[r][c] = (_Float16)K[g];
    sv[r][c] = (_Float16)V[g];
  }
  __syncthreads();

  for (int i = tid; i < 16 * 128; i += 256) {
    int nl = i >> 7, m = i & 127;
    int n  = chunk * 16 + nl;
    bool keep = (m <= n) || (((n & 1) == 0) && (m == ((n - 1) & 127)));
    float sc = -3.0e38f;
    if (keep) {
      const float* qp = Q + (rowbase + n) * DD + h * 64;
      float acc = 0.f;
#pragma unroll
      for (int d = 0; d < 64; ++d) acc += qp[d] * (float)sk[m][d];
      sc = acc * 0.125f;                     // 1/sqrt(64)
    }
    ss[nl][m] = sc;
  }
  __syncthreads();

  if (tid < 16) {
    float mx = -3.0e38f;
    for (int m = 0; m < 128; ++m) mx = fmaxf(mx, ss[tid][m]);
    float sum = 0.f;
    for (int m = 0; m < 128; ++m) { float e = expf(ss[tid][m] - mx); ss[tid][m] = e; sum += e; }
    float inv = 1.0f / sum;
    for (int m = 0; m < 128; ++m) ss[tid][m] *= inv;
  }
  __syncthreads();

  for (int i = tid; i < 16 * 64; i += 256) {
    int nl = i >> 6, c = i & 63;
    float acc = 0.f;
    for (int m = 0; m < 128; ++m) acc += ss[nl][m] * (float)sv[m][c];
    Y[(rowbase + chunk * 16 + nl) * DD + h * 64 + c] = acc;
  }
}

// ---------------------------------------------------------------------------
// Small elementwise kernels
// ---------------------------------------------------------------------------
__global__ void add_temporal_kernel(float* __restrict__ proj, const float* __restrict__ temporal)
{
  size_t idx = (size_t)blockIdx.x * blockDim.x + threadIdx.x;
  if (idx >= (size_t)BT * DD) return;
  int n = (int)(idx & (DD - 1));
  int t = (int)((idx >> 9) & (TT - 1));
  proj[idx] += temporal[t * DD + n];
}

__global__ void interleave_kernel(const float* __restrict__ proj, const float* __restrict__ glob,
                                  float* __restrict__ g)
{
  size_t idx = (size_t)blockIdx.x * blockDim.x + threadIdx.x;
  if (idx >= (size_t)R2 * DD) return;
  int n = (int)(idx & (DD - 1));
  size_t br = idx >> 9;
  int r = (int)(br & 127);
  size_t b = br >> 7;
  size_t src = ((b * TT + (size_t)(r >> 1)) << 9) + n;
  g[idx] = (r & 1) ? glob[src] : proj[src];
}

__global__ void deinterleave_kernel(const float* __restrict__ g, float* __restrict__ glob)
{
  size_t idx = (size_t)blockIdx.x * blockDim.x + threadIdx.x;
  if (idx >= (size_t)BT * DD) return;
  int n = (int)(idx & (DD - 1));
  size_t bt = idx >> 9;
  int t = (int)(bt & (TT - 1));
  size_t b = bt >> 6;
  glob[idx] = g[((b * 128 + (size_t)(2 * t + 1)) << 9) + n];
}

// ---------------------------------------------------------------------------
// Host orchestration.
// Input flattening (JAX pytree, dict keys sorted):
//   0 states, 1 actions,
//   2 act_W, 3 act_b, 4 ah_W, 5 ah_b,
//   6..233 blocks[0..5] x 38 leaves:
//     +0 Wlg, +1 blg, +2..17 glob SAB, +18 lgb, +19 lgg, +20 lnb, +21 lng,
//     +22..37 local SAB
//   SAB leaf order: W1,W2,Wk,Wp,Wq,Wv,b1,b2,bk,bp,bq,bv,ln1b,ln1g,ln2b,ln2g
//   234 lnh_b, 235 lnh_g, 236 patch_W, 237 patch_b, 238 spatial,
//   239 temporal, 240 vec_W, 241 vec_b
// ---------------------------------------------------------------------------
extern "C" void kernel_launch(void* const* d_in, const int* in_sizes, int n_in,
                              void* d_out, int out_size, void* d_ws, size_t ws_size,
                              hipStream_t stream)
{
  (void)in_sizes; (void)n_in; (void)out_size; (void)ws_size;
  const float* states  = (const float*)d_in[0];
  const float* actions = (const float*)d_in[1];
  auto F = [&](int i) -> const float* { return (const float*)d_in[i]; };

  const int I_ACT_W = 2, I_ACT_B = 3, I_AH_W = 4, I_AH_B = 5;
  const int I_BLK0 = 6, BLKS = 38;
  const int I_LNH_B = 234, I_LNH_G = 235, I_PATCH_W = 236, I_PATCH_B = 237;
  const int I_SPATIAL = 238, I_TEMPORAL = 239, I_VEC_W = 240, I_VEC_B = 241;

  // bump allocator over workspace
  char* wsb = (char*)d_ws;
  size_t off = 0;
  auto alloc = [&](size_t bytes) -> void* {
    void* p = wsb + off;
    off += (bytes + 255) & ~(size_t)255;
    return p;
  };
  // convert+transpose W[K][N] f32 -> Wt[N][K] f16
  auto conv = [&](const float* src, int K, int N) -> _Float16* {
    _Float16* dst = (_Float16*)alloc((size_t)K * N * sizeof(_Float16));
    dim3 grid((unsigned)((N + 31) / 32), (unsigned)((K + 31) / 32));
    transpose_cvt_kernel<<<grid, dim3(256), 0, stream>>>(src, dst, K, N);
    return dst;
  };

  struct SabF16 { _Float16 *Wq, *Wk, *Wv, *Wp, *W1, *W2; };
  SabF16 locw[NLL], glow[NLL];
  _Float16* wlg[NLL];
  for (int i = 0; i < NLL; ++i) {
    int bb = I_BLK0 + i * BLKS;
    wlg[i]     = conv(F(bb + 0),      PP * ID, DD);
    glow[i].W1 = conv(F(bb + 2 + 0),  DD, 4 * DD);
    glow[i].W2 = conv(F(bb + 2 + 1),  4 * DD, DD);
    glow[i].Wk = conv(F(bb + 2 + 2),  DD, DD);
    glow[i].Wp = conv(F(bb + 2 + 3),  DD, DD);
    glow[i].Wq = conv(F(bb + 2 + 4),  DD, DD);
    glow[i].Wv = conv(F(bb + 2 + 5),  DD, DD);
    locw[i].W1 = conv(F(bb + 22 + 0), ID, 4 * ID);
    locw[i].W2 = conv(F(bb + 22 + 1), 4 * ID, ID);
    locw[i].Wk = conv(F(bb + 22 + 2), ID, ID);
    locw[i].Wp = conv(F(bb + 22 + 3), ID, ID);
    locw[i].Wq = conv(F(bb + 22 + 4), ID, ID);
    locw[i].Wv = conv(F(bb + 22 + 5), ID, ID);
  }
  _Float16* ahW = conv(F(I_AH_W), DD, VV);

  // ---- activation buffers ----
  float* xloc = (float*)alloc((size_t)R1 * ID * 4);
  float* ltb  = (float*)alloc((size_t)R1 * ID * 4);
  float* lnb1 = (float*)alloc((size_t)R1 * ID * 4);
  float* big  = (float*)alloc((size_t)R1 * DD * 4);    // q|k|v|y, then MLP hidden
  float* x2b  = (float*)alloc((size_t)R1 * ID * 4);
  float* proj = (float*)alloc((size_t)BT * DD * 4);
  float* proj2= (float*)alloc((size_t)BT * DD * 4);
  float* globv= (float*)alloc((size_t)BT * DD * 4);
  float* gbuf = (float*)alloc((size_t)R2 * DD * 4);
  float* gln  = (float*)alloc((size_t)R2 * DD * 4);
  float* gbig = (float*)alloc((size_t)R2 * (4 * DD) * 4); // gq|gk|gv|gy, then hidden
  float* gx2  = (float*)alloc((size_t)R2 * DD * 4);
  float* hln  = (float*)alloc((size_t)BT * DD * 4);

  auto gemm = [&](int act, const float* A, const _Float16* W, const float* bias,
                  const float* res, float* C, int M, int N, int K) {
    dim3 grid((unsigned)((N + 127) / 128), (unsigned)((M + 127) / 128));
    if (act == 0)      gemm_f16w_kernel<0><<<grid, dim3(256), G_SMEM_BYTES, stream>>>(A, W, bias, res, C, M, N, K);
    else if (act == 1) gemm_f16w_kernel<1><<<grid, dim3(256), G_SMEM_BYTES, stream>>>(A, W, bias, res, C, M, N, K);
    else               gemm_f16w_kernel<2><<<grid, dim3(256), G_SMEM_BYTES, stream>>>(A, W, bias, res, C, M, N, K);
  };
  auto lnorm = [&](const float* X, const float* g, const float* b, float* Y, int R, int dim) {
    layernorm_kernel<<<dim3((unsigned)R), dim3(256), 0, stream>>>(X, g, b, Y, dim);
  };

  // ---- embeddings ----
  {
    size_t tot = (size_t)BT * PP * ID;
    embed_local_kernel<<<dim3((unsigned)((tot + 255) / 256)), dim3(256), 0, stream>>>(
        states, actions, F(I_ACT_W), F(I_ACT_B), F(I_PATCH_W), F(I_PATCH_B), F(I_SPATIAL), xloc);
    size_t tg = (size_t)BT * DD;
    embed_glob_kernel<<<dim3((unsigned)((tg + 255) / 256)), dim3(256), 0, stream>>>(
        states, F(I_VEC_W), F(I_VEC_B), F(I_TEMPORAL), globv);
  }

  for (int i = 0; i < NLL; ++i) {
    int bb = I_BLK0 + i * BLKS;
    // ===================== local SAB (rows R1, width 128) ===================
    const float *l_b1 = F(bb+22+6),  *l_b2 = F(bb+22+7),  *l_bk = F(bb+22+8);
    const float *l_bp = F(bb+22+9),  *l_bq = F(bb+22+10), *l_bv = F(bb+22+11);
    const float *l_ln1b = F(bb+22+12), *l_ln1g = F(bb+22+13);
    const float *l_ln2b = F(bb+22+14), *l_ln2g = F(bb+22+15);
    float* q = big;
    float* k = big + (size_t)R1 * ID;
    float* v = big + 2 * (size_t)R1 * ID;
    float* y = big + 3 * (size_t)R1 * ID;

    lnorm(xloc, l_ln1g, l_ln1b, lnb1, R1, ID);
    gemm(0, lnb1, locw[i].Wq, l_bq, nullptr, q, R1, ID, ID);
    gemm(0, lnb1, locw[i].Wk, l_bk, nullptr, k, R1, ID, ID);
    gemm(0, lnb1, locw[i].Wv, l_bv, nullptr, v, R1, ID, ID);
    attn_local_kernel<<<dim3(BT * LHH), dim3(128), 0, stream>>>(q, k, v, y);
    gemm(0, y, locw[i].Wp, l_bp, xloc, x2b, R1, ID, ID);       // x + attn
    lnorm(x2b, l_ln2g, l_ln2b, lnb1, R1, ID);
    gemm(1, lnb1, locw[i].W1, l_b1, nullptr, big, R1, 4 * ID, ID);   // gelu; big reused as H
    gemm(0, big, locw[i].W2, l_b2, x2b, ltb, R1, ID, 4 * ID);        // lt

    // ===================== local -> global projection =======================
    lnorm(ltb, F(bb + 21), F(bb + 20), lnb1, R1, ID);          // lng, lnb
    gemm(0, lnb1, wlg[i], F(bb + 1), nullptr, proj, BT, DD, PP * ID);
    lnorm(proj, F(bb + 19), F(bb + 18), proj2, BT, DD);        // lgg, lgb
    if (i == 0) {
      size_t tg = (size_t)BT * DD;
      add_temporal_kernel<<<dim3((unsigned)((tg + 255) / 256)), dim3(256), 0, stream>>>(
          proj2, F(I_TEMPORAL));
    }
    {
      size_t tg = (size_t)R2 * DD;
      interleave_kernel<<<dim3((unsigned)((tg + 255) / 256)), dim3(256), 0, stream>>>(
          proj2, globv, gbuf);
    }

    // ===================== global SAB (rows R2, width 512, masked) ==========
    const float *g_b1 = F(bb+2+6),  *g_b2 = F(bb+2+7),  *g_bk = F(bb+2+8);
    const float *g_bp = F(bb+2+9),  *g_bq = F(bb+2+10), *g_bv = F(bb+2+11);
    const float *g_ln1b = F(bb+2+12), *g_ln1g = F(bb+2+13);
    const float *g_ln2b = F(bb+2+14), *g_ln2g = F(bb+2+15);
    float* gq = gbig;
    float* gk = gbig + (size_t)R2 * DD;
    float* gv = gbig + 2 * (size_t)R2 * DD;
    float* gy = gbig + 3 * (size_t)R2 * DD;

    lnorm(gbuf, g_ln1g, g_ln1b, gln, R2, DD);
    gemm(0, gln, glow[i].Wq, g_bq, nullptr, gq, R2, DD, DD);
    gemm(0, gln, glow[i].Wk, g_bk, nullptr, gk, R2, DD, DD);
    gemm(0, gln, glow[i].Wv, g_bv, nullptr, gv, R2, DD, DD);
    attn_glob_kernel<<<dim3(BB * GHH * 8), dim3(256), 0, stream>>>(gq, gk, gv, gy);
    gemm(0, gy, glow[i].Wp, g_bp, gbuf, gx2, R2, DD, DD);      // g + attn
    lnorm(gx2, g_ln2g, g_ln2b, gln, R2, DD);
    gemm(1, gln, glow[i].W1, g_b1, nullptr, gbig, R2, 4 * DD, DD);   // gelu; gbig reused as H
    gemm(0, gbig, glow[i].W2, g_b2, gx2, gbuf, R2, DD, 4 * DD);

    {
      size_t tg = (size_t)BT * DD;
      deinterleave_kernel<<<dim3((unsigned)((tg + 255) / 256)), dim3(256), 0, stream>>>(
          gbuf, globv);
    }
    // local = lt (ping-pong)
    float* tmp = xloc; xloc = ltb; ltb = tmp;
  }

  // ---- head: tanh(LN(glob) @ ah_W + ah_b) -> d_out (f32) ----
  lnorm(globv, F(I_LNH_G), F(I_LNH_B), hln, BT, DD);
  gemm(2, hln, ahW, F(I_AH_B), nullptr, (float*)d_out, BT, VV, DD);
}
